// DeformableTransformerDecoderLayer_33285996544313
// MI455X (gfx1250) — compile-verified, matrix operand order checked by reading, not run on hardware
//
#include <hip/hip_runtime.h>

// ---------------------------------------------------------------------------
// DeformableTransformerDecoderLayer on gfx1250 (MI455X), bf16 WMMA pipeline
// with TDM (tensor_load_to_lds) staging of weight panels into LDS.
// BS=8, NQ=900, D=256, H=8, HD=32, LEN_V=13294, DFFN=1024, L=4, P=4
// ---------------------------------------------------------------------------

typedef __attribute__((ext_vector_type(16))) __bf16 v16bf;
typedef __attribute__((ext_vector_type(8)))  __bf16 v8bf;
typedef __attribute__((ext_vector_type(8)))  float  v8f;
typedef __attribute__((ext_vector_type(4)))  unsigned int u32x4;
typedef __attribute__((ext_vector_type(8)))  int i32x8;
typedef __attribute__((ext_vector_type(4)))  int i32x4;

#define BS      8
#define NQ      900
#define DMODEL  256
#define NHEADS  8
#define HD      32
#define LEN_V   13294
#define DFFN    1024
#define M_TOK   (BS*NQ)        // 7200
#define M_VAL   (BS*LEN_V)     // 106352
#define QTILES  57             // ceil(900/16)
#define KCHUNKS 29             // ceil(900/32)
#define VT_STRIDE 928          // 29*32, padded key stride for V^T

// Combine two 16B bf16 chunks into a WMMA 16-element operand.
__device__ __forceinline__ v16bf load_ab(const __bf16* p0, const __bf16* p1) {
  v8bf lo = *(const v8bf*)p0;
  v8bf hi = *(const v8bf*)p1;
  return __builtin_shufflevector(lo, hi, 0,1,2,3,4,5,6,7,8,9,10,11,12,13,14,15);
}

// ---------------------------------------------------------------------------
// Panel GEMM: out[M,N] = A[M,K] @ Bt[N,K]^T + bias (+resid) (ReLU?)
// Block (8 waves) owns a 128(M) x 64(N) output panel. The 64xK weight panel
// is staged 64-K-elements at a time into LDS by the Tensor Data Mover;
// each wave computes one 16-row strip x four 16-col tiles (8 WMMA / chunk).
// Requires: N % 64 == 0, K % 64 == 0. M ragged rows handled by clamp+guard.
// ---------------------------------------------------------------------------
template<bool OUT_BF16, bool RELU, bool HAS_RES>
__global__ __launch_bounds__(256) void gemm_bf16_kernel(
    const __bf16* __restrict__ A, const __bf16* __restrict__ Bt,
    const float* __restrict__ bias, const float* __restrict__ resid,
    void* __restrict__ out, int M, int N, int K)
{
  __shared__ __bf16 sB[64 * 64];                 // 8 KB weight panel chunk
  int lane = threadIdx.x & 31;
  int wv   = threadIdx.x >> 5;
  int panels = N >> 6;
  int bm = blockIdx.x / panels;
  int pn = blockIdx.x % panels;
  int n0 = pn * 64;
  int l15 = lane & 15;
  int klo = (lane < 16) ? 0 : 8;
  int row  = bm*128 + wv*16 + l15;
  int rowc = row < (M-1) ? row : (M-1);
  const __bf16* Ap = A + (long)rowc * K + klo;
  unsigned ldsoff = (unsigned)(unsigned long long)(void*)&sB[0];

  v8f acc[4] = {};
  for (int k0 = 0; k0 < K; k0 += 64) {
    __syncthreads();                              // previous chunk consumed
    if (wv == 0) {
      // TDM descriptor: 2D tile, 64 rows (dim1) x 64 elems (dim0), bf16.
      unsigned long long ga =
          (unsigned long long)(const void*)(Bt + (long)n0 * K + k0);
      u32x4 g0 = { 1u,                                   // count=1 (valid)
                   ldsoff,                               // lds_addr
                   (unsigned)(ga & 0xFFFFFFFFu),
                   (unsigned)((ga >> 32) & 0x1FFFFFFu) | (2u << 30) }; // type=2
      i32x8 g1 = { (int)(1u << 16),                      // data_size=1 (2B)
                   (int)(((unsigned)K & 0xFFFFu) << 16), // tensor_dim0 lo
                   (int)((((unsigned)K >> 16) & 0xFFFFu) |
                         (((unsigned)N & 0xFFFFu) << 16)), // td0 hi | td1 lo
                   (int)((((unsigned)N >> 16) & 0xFFFFu) |
                         (64u << 16)),                   // td1 hi | tile_dim0=64
                   (int)64,                              // tile_dim1=64, tile_dim2=0
                   (int)K,                               // tensor_dim0_stride lo
                   0, 0 };
      i32x4 gz4 = {0, 0, 0, 0};
      i32x8 gz8 = {0, 0, 0, 0, 0, 0, 0, 0};
      __builtin_amdgcn_tensor_load_to_lds(g0, g1, gz4, gz4, gz8, 0);
      __builtin_amdgcn_s_wait_tensorcnt(0);
    }
    __syncthreads();                              // panel chunk visible
#pragma unroll
    for (int kk = 0; kk < 64; kk += 32) {
      v16bf a = load_ab(Ap + k0 + kk, Ap + k0 + kk + 16);
#pragma unroll
      for (int j = 0; j < 4; j++) {
        const __bf16* bp = &sB[(j*16 + l15)*64 + kk + klo];
        v16bf b = load_ab(bp, bp + 16);
        acc[j] = __builtin_amdgcn_wmma_f32_16x16x32_bf16(false, a, false, b,
                                                         (short)0, acc[j],
                                                         false, false);
      }
    }
  }
  int ro = (lane < 16) ? 0 : 8;
#pragma unroll
  for (int j = 0; j < 4; j++) {
    int n = n0 + j*16 + l15;
    float bn = bias[n];
#pragma unroll
    for (int r = 0; r < 8; r++) {
      long m = (long)bm*128 + wv*16 + r + ro;
      if (m < M) {
        float v = acc[j][r] + bn;
        if (RELU)    v = v > 0.f ? v : 0.f;
        if (HAS_RES) v += resid[m * N + n];
        if (OUT_BF16) ((__bf16*)out)[m * N + n] = (__bf16)v;
        else          ((float*)out) [m * N + n] = v;
      }
    }
  }
}

// ---------------------------------------------------------------------------
// Flash self-attention (one wave = one 16-query tile of one (b,h)).
// Computes S^T = K x Q^T so per-query softmax stats live per-lane, and the
// probability tile feeds the P*V WMMA as B-operand via per-lane packing.
// ---------------------------------------------------------------------------
__global__ __launch_bounds__(256) void flash_attn_kernel(
    const __bf16* __restrict__ qb, const __bf16* __restrict__ kb,
    const __bf16* __restrict__ vt, __bf16* __restrict__ attn_out)
{
  int lane = threadIdx.x & 31;
  int wid  = blockIdx.x * 8 + (threadIdx.x >> 5);
  int qt = wid % QTILES;
  int h  = (wid / QTILES) & 7;
  int b  = wid / (QTILES * NHEADS);
  int l15 = lane & 15;
  int klo = (lane < 16) ? 0 : 8;

  int qrow = qt*16 + l15; qrow = qrow < (NQ-1) ? qrow : (NQ-1);
  const __bf16* qp = qb + ((long)(b*NQ + qrow))*DMODEL + h*HD + klo;
  v16bf Bq = load_ab(qp, qp + 16);

  v8f o0 = {}, o1 = {};
  float mrun = -3.0e38f, lrun = 0.f;
  const float scale = 0.17677669529663687f;   // 1/sqrt(32)
  const __bf16* vbase = vt + ((long)((b*NHEADS + h)*HD)) * VT_STRIDE;

  for (int kc = 0; kc < KCHUNKS; kc++) {
    int k0 = kc * 32;
    v8f s[2];
#pragma unroll
    for (int t = 0; t < 2; t++) {
      int krow = k0 + t*16 + l15; krow = krow < (NQ-1) ? krow : (NQ-1);
      const __bf16* kp = kb + ((long)(b*NQ + krow))*DMODEL + h*HD + klo;
      v16bf Ak = load_ab(kp, kp + 16);
      v8f zc = {};
      s[t] = __builtin_amdgcn_wmma_f32_16x16x32_bf16(false, Ak, false, Bq,
                                                     (short)0, zc, false, false);
    }
    float mloc = -3.0e38f;
#pragma unroll
    for (int t = 0; t < 2; t++)
#pragma unroll
      for (int r = 0; r < 8; r++) {
        int key = k0 + t*16 + r + klo;           // C-layout row = key index
        float sv = (key < NQ) ? s[t][r] * scale : -3.0e38f;
        s[t][r] = sv;
        mloc = fmaxf(mloc, sv);
      }
    mloc = fmaxf(mloc, __shfl_xor(mloc, 16, 32));
    float mnew  = fmaxf(mrun, mloc);
    float alpha = __expf(mrun - mnew);
    float lloc = 0.f;
    v16bf bp;
#pragma unroll
    for (int t = 0; t < 2; t++)
#pragma unroll
      for (int r = 0; r < 8; r++) {
        float p = __expf(s[t][r] - mnew);
        lloc += p;
        bp[t*8 + r] = (__bf16)p;                 // P^T as B-operand, no shuffle
      }
    lloc += __shfl_xor(lloc, 16, 32);
    lrun = lrun * alpha + lloc;
    mrun = mnew;
#pragma unroll
    for (int r = 0; r < 8; r++) { o0[r] *= alpha; o1[r] *= alpha; }

    const __bf16* vp0 = vbase + (long)l15        * VT_STRIDE + k0 + klo;
    const __bf16* vp1 = vbase + (long)(l15 + 16) * VT_STRIDE + k0 + klo;
    v16bf Av0 = load_ab(vp0, vp0 + 16);
    v16bf Av1 = load_ab(vp1, vp1 + 16);
    o0 = __builtin_amdgcn_wmma_f32_16x16x32_bf16(false, Av0, false, bp,
                                                 (short)0, o0, false, false);
    o1 = __builtin_amdgcn_wmma_f32_16x16x32_bf16(false, Av1, false, bp,
                                                 (short)0, o1, false, false);
  }
  float inv = 1.f / lrun;
  int q = qt*16 + l15;
  if (q < NQ) {
    int ro = (lane < 16) ? 0 : 8;
    __bf16* op = attn_out + ((long)(b*NQ + q))*DMODEL + h*HD;
#pragma unroll
    for (int r = 0; r < 8; r++) {
      op[ro + r]      = (__bf16)(o0[r] * inv);
      op[ro + r + 16] = (__bf16)(o1[r] * inv);
    }
  }
}

// ---------------------------------------------------------------------------
// Deformable attention: 1 wave = (b,q,h); lane = head-dim channel.
// ---------------------------------------------------------------------------
__global__ __launch_bounds__(256) void deform_sample_kernel(
    const __bf16* __restrict__ val, const float* __restrict__ off,
    const float* __restrict__ aw,  const float* __restrict__ rbbox,
    __bf16* __restrict__ outb)
{
  int lane = threadIdx.x & 31;                       // hd
  int wid  = blockIdx.x * 8 + (threadIdx.x >> 5);
  int h  = wid & 7;
  int bq = wid >> 3;
  int b  = bq / NQ;
  const float* rbp  = rbbox + (long)bq * 4;
  float cx = rbp[0], cy = rbp[1], bw = rbp[2], bh = rbp[3];
  const float* offp = off + (long)bq * DMODEL + h * 32;
  const float* awp  = aw  + (long)bq * 128    + h * 16;
  const int Hs[4] = {100,50,25,13};
  const int St[4] = {0,10000,12500,13125};
  const __bf16* vrow = val + ((long)b * LEN_V) * DMODEL + h*HD + lane;
  float acc = 0.f;
#pragma unroll
  for (int l = 0; l < 4; l++) {
    int Hg = Hs[l], Wg = Hs[l];
    long base = (long)St[l] * DMODEL;
#pragma unroll
    for (int p = 0; p < 4; p++) {
      float ox = offp[l*8 + p*2 + 0];
      float oy = offp[l*8 + p*2 + 1];
      float a  = awp [l*4 + p];
      float lx = cx + ox * 0.25f * bw * 0.5f;
      float ly = cy + oy * 0.25f * bh * 0.5f;
      float x = lx * (float)Wg - 0.5f;
      float y = ly * (float)Hg - 0.5f;
      float x0f = floorf(x), y0f = floorf(y);
      float fx = x - x0f,   fy = y - y0f;
      int x0 = (int)x0f, y0 = (int)y0f;
#pragma unroll
      for (int dy = 0; dy < 2; dy++)
#pragma unroll
        for (int dx = 0; dx < 2; dx++) {
          int xi = x0 + dx, yi = y0 + dy;
          bool ok = (xi >= 0) & (xi < Wg) & (yi >= 0) & (yi < Hg);
          int xc = xi < 0 ? 0 : (xi > Wg-1 ? Wg-1 : xi);
          int yc = yi < 0 ? 0 : (yi > Hg-1 ? Hg-1 : yi);
          float wgt = (dx ? fx : 1.f-fx) * (dy ? fy : 1.f-fy);
          wgt = ok ? wgt : 0.f;
          float g = (float)vrow[base + (long)(yc*Wg + xc) * DMODEL];
          acc += g * wgt * a;
        }
    }
  }
  outb[(long)bq * DMODEL + h*HD + lane] = (__bf16)acc;
}

// ---------------------------------------------------------------------------
// LayerNorm over 256-wide rows; 1 wave per row. Optional bf16 copy and
// optional bf16(x + query_pos) copy for the next stage.
// ---------------------------------------------------------------------------
__global__ __launch_bounds__(256) void layernorm_kernel(
    const float* __restrict__ x, const float* __restrict__ g,
    const float* __restrict__ bt, float* __restrict__ yf,
    __bf16* __restrict__ yb, const float* __restrict__ addpos,
    __bf16* __restrict__ ypb)
{
  int lane = threadIdx.x & 31;
  long row = (long)blockIdx.x * 8 + (threadIdx.x >> 5);
  const float* xp = x + row * DMODEL;
  float v[8];
#pragma unroll
  for (int i = 0; i < 8; i++) v[i] = xp[lane + i*32];
  float s = 0.f;
#pragma unroll
  for (int i = 0; i < 8; i++) s += v[i];
  for (int o = 16; o; o >>= 1) s += __shfl_xor(s, o, 32);
  float mean = s * (1.f/256.f);
  float vs = 0.f;
#pragma unroll
  for (int i = 0; i < 8; i++) { float d = v[i] - mean; vs += d*d; }
  for (int o = 16; o; o >>= 1) vs += __shfl_xor(vs, o, 32);
  float rstd = rsqrtf(vs * (1.f/256.f) + 1e-5f);
#pragma unroll
  for (int i = 0; i < 8; i++) {
    int n = lane + i*32;
    float yv = (v[i] - mean) * rstd * g[n] + bt[n];
    if (yf)  yf[row*DMODEL + n] = yv;
    if (yb)  yb[row*DMODEL + n] = (__bf16)yv;
    if (ypb) ypb[row*DMODEL + n] = (__bf16)(yv + addpos[row*DMODEL + n]);
  }
}

// ---------------------------------------------------------------------------
// Small utility kernels
// ---------------------------------------------------------------------------
__global__ __launch_bounds__(256) void cast_bf16_kernel(
    const float* __restrict__ x, __bf16* __restrict__ y, long n)
{
  long i = (long)blockIdx.x * 256 + threadIdx.x;
  if (i < n) y[i] = (__bf16)x[i];
}

__global__ __launch_bounds__(256) void add_cast_bf16_kernel(
    const float* __restrict__ a, const float* __restrict__ b,
    __bf16* __restrict__ y, long n)
{
  long i = (long)blockIdx.x * 256 + threadIdx.x;
  if (i < n) y[i] = (__bf16)(a[i] + b[i]);
}

__global__ __launch_bounds__(256) void transpose_cast_kernel(
    const float* __restrict__ W, __bf16* __restrict__ Wt, int K, int N)
{
  long i = (long)blockIdx.x * 256 + threadIdx.x;
  if (i >= (long)K * N) return;
  int n = (int)(i % N);
  long k = i / N;
  Wt[(long)n * K + k] = (__bf16)W[i];
}

// v_bf [b*900+m][h*32+hd] -> vt [((b*8+h)*32+hd)*928 + m], zero padded keys
__global__ __launch_bounds__(256) void v_transpose_kernel(
    const __bf16* __restrict__ v, __bf16* __restrict__ vt)
{
  long tid = (long)blockIdx.x * 256 + threadIdx.x;
  if (tid >= (long)BS*NHEADS*HD*VT_STRIDE) return;
  int m = (int)(tid % VT_STRIDE);
  long r = tid / VT_STRIDE;
  int hd = (int)(r & 31);
  int h  = (int)((r >> 5) & 7);
  int b  = (int)(r >> 8);
  __bf16 out = (__bf16)0.f;
  if (m < NQ) out = v[((long)(b*NQ + m))*DMODEL + h*HD + hd];
  vt[tid] = out;
}

// softmax over groups of 16 (L*P) — two groups per wave, in-place
__global__ __launch_bounds__(256) void aw_softmax_kernel(float* __restrict__ aw)
{
  int lane = threadIdx.x & 31;
  long wv = (long)blockIdx.x * 8 + (threadIdx.x >> 5);
  long gidx = wv*2 + (lane >> 4);
  float s = aw[gidx*16 + (lane & 15)];
  float m = s;
  for (int o = 1; o < 16; o <<= 1) m = fmaxf(m, __shfl_xor(m, o, 32));
  float e = __expf(s - m);
  float t = e;
  for (int o = 1; o < 16; o <<= 1) t += __shfl_xor(t, o, 32);
  aw[gidx*16 + (lane & 15)] = e / t;
}

// ---------------------------------------------------------------------------
// Host launcher
// ---------------------------------------------------------------------------
extern "C" void kernel_launch(void* const* d_in, const int* in_sizes, int n_in,
                              void* d_out, int out_size, void* d_ws, size_t ws_size,
                              hipStream_t stream) {
  const float* embed  = (const float*)d_in[0];
  const float* rbbox  = (const float*)d_in[1];
  const float* feats  = (const float*)d_in[2];
  const float* qpos   = (const float*)d_in[3];
  const float* Wq = (const float*)d_in[4];  const float* bq = (const float*)d_in[5];
  const float* Wk = (const float*)d_in[6];  const float* bk = (const float*)d_in[7];
  const float* Wv = (const float*)d_in[8];  const float* bv = (const float*)d_in[9];
  const float* Wo = (const float*)d_in[10]; const float* bo = (const float*)d_in[11];
  const float* g1 = (const float*)d_in[12]; const float* b1 = (const float*)d_in[13];
  const float* Wval = (const float*)d_in[14]; const float* bval = (const float*)d_in[15];
  const float* Woff = (const float*)d_in[16]; const float* boff = (const float*)d_in[17];
  const float* Waw  = (const float*)d_in[18]; const float* baw  = (const float*)d_in[19];
  const float* Wop  = (const float*)d_in[20]; const float* bop  = (const float*)d_in[21];
  const float* g2 = (const float*)d_in[22]; const float* b2 = (const float*)d_in[23];
  const float* W1 = (const float*)d_in[24]; const float* bf1 = (const float*)d_in[25];
  const float* W2 = (const float*)d_in[26]; const float* bf2 = (const float*)d_in[27];
  const float* g3 = (const float*)d_in[28]; const float* b3 = (const float*)d_in[29];

  char* wsp = (char*)d_ws;
  auto alloc = [&](size_t bytes) -> void* {
    void* p = (void*)wsp;
    wsp += (bytes + 255) & ~(size_t)255;
    return p;
  };
  const long nTok = (long)M_TOK * DMODEL;            // 1,843,200
  const long nVal = (long)M_VAL * DMODEL;            // 27,226,112

  __bf16* qkin_bf  = (__bf16*)alloc(nTok * 2);
  __bf16* emb_bf   = (__bf16*)alloc(nTok * 2);
  __bf16* q_bf     = (__bf16*)alloc(nTok * 2);
  __bf16* k_bf     = (__bf16*)alloc(nTok * 2);
  __bf16* v_bf     = (__bf16*)alloc(nTok * 2);
  __bf16* vt_bf    = (__bf16*)alloc((long)BS*NHEADS*HD*VT_STRIDE * 2);
  __bf16* attn_bf  = (__bf16*)alloc(nTok * 2);
  float*  res_f    = (float*) alloc(nTok * 4);       // reused for all 3 residual sums
  float*  e1_f     = (float*) alloc(nTok * 4);
  __bf16* q2_bf    = (__bf16*)alloc(nTok * 2);
  __bf16* feats_bf = (__bf16*)alloc(nVal * 2);       // reused as FFN hidden (bf16)
  __bf16* val_bf   = (__bf16*)alloc(nVal * 2);
  float*  off_f    = (float*) alloc(nTok * 4);
  float*  aw_f     = (float*) alloc((long)M_TOK * 128 * 4);
  __bf16* ca_bf    = (__bf16*)alloc(nTok * 2);
  float*  e2_f     = (float*) alloc(nTok * 4);
  __bf16* e2_bf    = (__bf16*)alloc(nTok * 2);
  __bf16* Wqt  = (__bf16*)alloc(256*256*2);
  __bf16* Wkt  = (__bf16*)alloc(256*256*2);
  __bf16* Wvt  = (__bf16*)alloc(256*256*2);
  __bf16* Wot  = (__bf16*)alloc(256*256*2);
  __bf16* Wvalt= (__bf16*)alloc(256*256*2);
  __bf16* Wofft= (__bf16*)alloc(256*256*2);
  __bf16* Wawt = (__bf16*)alloc(256*128*2);
  __bf16* Wopt = (__bf16*)alloc(256*256*2);
  __bf16* W1t  = (__bf16*)alloc((long)256*1024*2);
  __bf16* W2t  = (__bf16*)alloc((long)1024*256*2);
  __bf16* h_bf = feats_bf;                           // alias: feats_bf dead by then

  dim3 blk(256);
  auto cdiv = [](long a, long b) { return (unsigned)((a + b - 1) / b); };

  // --- prepare activations ---
  add_cast_bf16_kernel<<<cdiv(nTok,256), blk, 0, stream>>>(embed, qpos, qkin_bf, nTok);
  cast_bf16_kernel   <<<cdiv(nTok,256), blk, 0, stream>>>(embed, emb_bf, nTok);
  cast_bf16_kernel   <<<cdiv(nVal,256), blk, 0, stream>>>(feats, feats_bf, nVal);

  // --- transpose+cast weights ---
  transpose_cast_kernel<<<cdiv(256*256,256), blk, 0, stream>>>(Wq,  Wqt,  256, 256);
  transpose_cast_kernel<<<cdiv(256*256,256), blk, 0, stream>>>(Wk,  Wkt,  256, 256);
  transpose_cast_kernel<<<cdiv(256*256,256), blk, 0, stream>>>(Wv,  Wvt,  256, 256);
  transpose_cast_kernel<<<cdiv(256*256,256), blk, 0, stream>>>(Wo,  Wot,  256, 256);
  transpose_cast_kernel<<<cdiv(256*256,256), blk, 0, stream>>>(Wval,Wvalt,256, 256);
  transpose_cast_kernel<<<cdiv(256*256,256), blk, 0, stream>>>(Woff,Wofft,256, 256);
  transpose_cast_kernel<<<cdiv(256*128,256), blk, 0, stream>>>(Waw, Wawt, 256, 128);
  transpose_cast_kernel<<<cdiv(256*256,256), blk, 0, stream>>>(Wop, Wopt, 256, 256);
  transpose_cast_kernel<<<cdiv((long)256*1024,256), blk, 0, stream>>>(W1, W1t, 256, 1024);
  transpose_cast_kernel<<<cdiv((long)1024*256,256), blk, 0, stream>>>(W2, W2t, 1024, 256);

  // --- self attention ---
  unsigned gP256 = cdiv(M_TOK,128) * (256/64);       // 57*4 = 228 blocks
  gemm_bf16_kernel<true,false,false><<<gP256, blk, 0, stream>>>(qkin_bf, Wqt, bq, nullptr, q_bf, M_TOK, 256, 256);
  gemm_bf16_kernel<true,false,false><<<gP256, blk, 0, stream>>>(qkin_bf, Wkt, bk, nullptr, k_bf, M_TOK, 256, 256);
  gemm_bf16_kernel<true,false,false><<<gP256, blk, 0, stream>>>(emb_bf,  Wvt, bv, nullptr, v_bf, M_TOK, 256, 256);
  v_transpose_kernel<<<cdiv((long)BS*NHEADS*HD*VT_STRIDE,256), blk, 0, stream>>>(v_bf, vt_bf);
  flash_attn_kernel<<<cdiv((long)BS*NHEADS*QTILES,8), blk, 0, stream>>>(q_bf, k_bf, vt_bf, attn_bf);
  gemm_bf16_kernel<false,false,true><<<gP256, blk, 0, stream>>>(attn_bf, Wot, bo, embed, res_f, M_TOK, 256, 256);
  layernorm_kernel<<<cdiv(M_TOK,8), blk, 0, stream>>>(res_f, g1, b1, e1_f, nullptr, qpos, q2_bf);

  // --- deformable cross attention ---
  unsigned gVal = cdiv(M_VAL,128) * (256/64);        // 831*4 = 3324 blocks
  gemm_bf16_kernel<true,false,false><<<gVal, blk, 0, stream>>>(feats_bf, Wvalt, bval, nullptr, val_bf, M_VAL, 256, 256);
  gemm_bf16_kernel<false,false,false><<<gP256, blk, 0, stream>>>(q2_bf, Wofft, boff, nullptr, off_f, M_TOK, 256, 256);
  unsigned gAw = cdiv(M_TOK,128) * (128/64);         // 57*2 = 114 blocks
  gemm_bf16_kernel<false,false,false><<<gAw, blk, 0, stream>>>(q2_bf, Wawt, baw, nullptr, aw_f, M_TOK, 128, 256);
  aw_softmax_kernel<<<cdiv((long)M_TOK*NHEADS/2,8), blk, 0, stream>>>(aw_f);
  deform_sample_kernel<<<cdiv((long)M_TOK*NHEADS,8), blk, 0, stream>>>(val_bf, off_f, aw_f, rbbox, ca_bf);
  gemm_bf16_kernel<false,false,true><<<gP256, blk, 0, stream>>>(ca_bf, Wopt, bop, e1_f, res_f, M_TOK, 256, 256);
  layernorm_kernel<<<cdiv(M_TOK,8), blk, 0, stream>>>(res_f, g2, b2, e2_f, e2_bf, nullptr, nullptr);

  // --- FFN ---
  unsigned gF1 = cdiv(M_TOK,128) * (DFFN/64);        // 57*16 = 912 blocks
  gemm_bf16_kernel<true,true,false><<<gF1, blk, 0, stream>>>(e2_bf, W1t, bf1, nullptr, h_bf, M_TOK, DFFN, 256);
  gemm_bf16_kernel<false,false,true><<<gP256, blk, 0, stream>>>(h_bf, W2t, bf2, e2_f, res_f, M_TOK, 256, DFFN);
  layernorm_kernel<<<cdiv(M_TOK,8), blk, 0, stream>>>(res_f, g3, b3, (float*)d_out, nullptr, nullptr, nullptr);
}
